// LSTM_53781580481228
// MI455X (gfx1250) — matrix operand order repaired and checked by reading, not compiled, not run on hardware
//
#include <hip/hip_runtime.h>
#include <cstdint>
#include <cstddef>

// Problem constants (fixed by the reference).
#define Bsz  4096
#define Lseq 60
#define Iin  13
#define Hh   128   // hidden
#define Gg   512   // 4*H gates
#define D2   256   // 2*H bidirectional feature
#define RT_N (Bsz / 16)   // 256 row tiles

typedef __bf16 bf16;
typedef __attribute__((ext_vector_type(16))) __bf16 v16bf;
typedef __attribute__((ext_vector_type(8)))  __bf16 v8bf;
typedef __attribute__((ext_vector_type(8)))  float  v8f;

// ---------------------------------------------------------------------------
// WMMA helpers (CDNA5 wave32, v_wmma_f32_16x16x32_bf16)
// ---------------------------------------------------------------------------
static __device__ __forceinline__ v8f wmma_bf16(v16bf a, v16bf b, v8f c) {
    // (neg_a, A, neg_b, B, c_mod, C, reuse_a, reuse_b)
    return __builtin_amdgcn_wmma_f32_16x16x32_bf16(false, a, false, b,
                                                   (short)0, c, false, false);
}

// A-matrix fragment, 16x32 bf16 (M x K). ISA 7.12.2:
//   lanes 0-15 hold row M=lane, K {0..7, 16..23}; lanes 16-31 hold row M=lane-16,
//   K {8..15, 24..31}. Pairs contiguous in K -> two 16B loads per lane.
static __device__ __forceinline__ v16bf load_frag_a(const bf16* base, int stride,
                                                    int row0, int k0) {
    int lane = threadIdx.x & 31;
    int m = lane & 15;
    int hi = lane >> 4;
    const bf16* p = base + (size_t)(row0 + m) * stride + k0 + hi * 8;
    v8bf lo = *(const v8bf*)(p);        // K +0..7
    v8bf hv = *(const v8bf*)(p + 16);   // K +16..23
    v16bf r;
#pragma unroll
    for (int i = 0; i < 8; ++i) { r[i] = lo[i]; r[8 + i] = hv[i]; }
    return r;
}

// B-matrix fragment, 32x16 bf16 (K x N): lane l holds column N=n0+(l&15),
//   K {0..7, 16..23} (lanes 0-15) / K {8..15, 24..31} (lanes 16-31).
static __device__ __forceinline__ v16bf load_frag_b(const bf16* base, int stride,
                                                    int k0, int n0) {
    int lane = threadIdx.x & 31;
    const bf16* p = base + (size_t)(k0 + (lane >> 4) * 8) * stride + n0 + (lane & 15);
    v16bf r;
#pragma unroll
    for (int i = 0; i < 8; ++i) { r[i] = p[(size_t)i * stride]; r[8 + i] = p[(size_t)(16 + i) * stride]; }
    return r;
}

// Fast activations: v_rcp_f32 + v_exp_f32 (single-instruction transcendentals).
static __device__ __forceinline__ float sigf(float x) {
    return __builtin_amdgcn_rcpf(1.0f + __expf(-x));
}
static __device__ __forceinline__ float tanhfast(float x) {
    // tanh(x) = 1 - 2/(exp(2x)+1)
    return 1.0f - 2.0f * __builtin_amdgcn_rcpf(__expf(2.0f * x) + 1.0f);
}

// Convert a bf16x8 fragment slice to f32 accumulator fragment.
static __device__ __forceinline__ v8f cvt8(v8bf v) {
    v8f r;
#pragma unroll
    for (int i = 0; i < 8; ++i) r[i] = (float)v[i];
    return r;
}

// xp is stored in C-fragment layout:
//   xp[(((t * RT_N + rowTile) * 32 + gateTile) * 256) + lane*8 + r]   (bf16)
// where rowTile = b/16, gateTile = n/16, and (lane, r) follow the ISA C-frag
// mapping: element (m = r + 8*(lane>>4), nn = lane&15).

// ---------------------------------------------------------------------------
// Weight transpose + f32 -> bf16 convert:  w[512][K] -> wt[K][512]
// ---------------------------------------------------------------------------
__global__ void wtrans_kernel(const float* __restrict__ w, bf16* __restrict__ wt,
                              int K) {
    int idx = blockIdx.x * 256 + threadIdx.x;
    if (idx >= K * Gg) return;
    int k = idx / Gg;
    int n = idx - k * Gg;
    wt[idx] = (bf16)w[(size_t)n * K + k];
}

// ---------------------------------------------------------------------------
// Layer-0 input projection (K = 13, VALU), writing frag-layout xp.
// ---------------------------------------------------------------------------
__global__ void xp0_kernel(const float* __restrict__ x,     // [B][L][13]
                           const float* __restrict__ w_ih,  // [512][13]
                           const float* __restrict__ b_ih,
                           const float* __restrict__ b_hh,
                           bf16* __restrict__ xp) {
    size_t idx = (size_t)blockIdx.x * 256 + threadIdx.x;
    int n = (int)(idx & (Gg - 1));
    size_t tb = idx >> 9;
    int b = (int)(tb & (Bsz - 1));
    int t = (int)(tb >> 12);
    if (t >= Lseq) return;
    const float* xr = x + ((size_t)b * Lseq + t) * Iin;
    const float* wr = w_ih + (size_t)n * Iin;
    float acc = b_ih[n] + b_hh[n];
#pragma unroll
    for (int k = 0; k < Iin; ++k) acc += xr[k] * wr[k];
    // scatter into fragment layout
    int rt2 = b >> 4, m = b & 15;
    int gt = n >> 4, nn = n & 15;
    int hi = m >> 3, r = m & 7;
    int lane = nn + 16 * hi;
    size_t fidx = ((((size_t)t * RT_N + rt2) * 32 + gt) * 256) + lane * 8 + r;
    xp[fidx] = (bf16)acc;
}

// ---------------------------------------------------------------------------
// Layers 1-2 input projection GEMM (WMMA, K = 256):
//   xp[t][b][n] = sum_k yin[t][b][k] * wihT[k][n] + b_ih[n] + b_hh[n]
// Block: 256 threads (8 waves), M-tile 64 rows for one timestep.
// wihT staged in LDS (256 KB, within 320 KB CDNA5 WGP LDS). Epilogue writes
// frag-layout bf16 xp with fully coalesced 16B vector stores.
// ---------------------------------------------------------------------------
__global__ __launch_bounds__(256)
void xp_gemm_kernel(const bf16* __restrict__ yin,   // [L][B][256]
                    const bf16* __restrict__ wihT,  // [256][512]
                    const float* __restrict__ b_ih,
                    const float* __restrict__ b_hh,
                    bf16* __restrict__ xp) {
    __shared__ __align__(32) bf16 sW[D2 * Gg];
    int tid = threadIdx.x, wid = tid >> 5, lane = tid & 31;
    int t = blockIdx.y;
    int b0 = blockIdx.x * 64;

    {   // stage w_ih^T in LDS (16B vector copies)
        const uint4* src = (const uint4*)wihT;
        uint4* dst = (uint4*)sW;
        for (int i = tid; i < (D2 * Gg) / 8; i += 256) dst[i] = src[i];
    }
    __syncthreads();

    int rt = wid & 3;            // row tile (16 rows)
    int ch = (wid >> 2) * 16;    // column-tile base (tiles [ch, ch+16))

    v8f zero = {};
    v8f acc[16];
#pragma unroll
    for (int c = 0; c < 16; ++c) acc[c] = zero;

    const bf16* arow = yin + ((size_t)t * Bsz + b0 + rt * 16) * D2;
#pragma unroll
    for (int kt = 0; kt < 8; ++kt) {
        v16bf A = load_frag_a(arow, D2, 0, kt * 32);
#pragma unroll
        for (int c = 0; c < 16; ++c) {
            v16bf Bf = load_frag_b(sW, Gg, kt * 32, (ch + c) * 16);
            acc[c] = wmma_bf16(A, Bf, acc[c]);
        }
    }

    int nn = lane & 15;
    int rtG = (b0 >> 4) + rt;    // global row tile
#pragma unroll
    for (int c = 0; c < 16; ++c) {
        int gt = ch + c;
        int n = gt * 16 + nn;
        float bias = b_ih[n] + b_hh[n];
        v8bf o;
#pragma unroll
        for (int r = 0; r < 8; ++r) o[r] = (bf16)(acc[c][r] + bias);
        size_t fidx = ((((size_t)t * RT_N + rtG) * 32 + gt) * 256) + lane * 8;
        *(v8bf*)(xp + fidx) = o;    // 16B/lane, 512B contiguous per wave
    }
}

// ---------------------------------------------------------------------------
// Recurrent LSTM scan (one direction of one layer).
// Block: 128 threads (4 waves), 64 batch rows; grid = B/64.
// w_hh^T (128x512 bf16, 128 KB) lives in LDS; each wave owns 16 batch rows,
// keeps cell state c in VGPRs for all 60 steps, and round-trips h through its
// private LDS slice (same-wave DS ops are in-order -> no barriers in t-loop).
// A-fragments of h(t-1) are hoisted into registers BEFORE any h(t) writes.
// xp enters through the WMMA C operand (frag-layout 16B vector loads).
// ---------------------------------------------------------------------------
__global__ __launch_bounds__(128)
void lstm_scan_kernel(const bf16* __restrict__ xp,    // frag-layout, incl. biases
                      bf16* __restrict__ yout,        // [L][B][256]
                      const bf16* __restrict__ whhT,  // [128][512]
                      int dirOff, int reverse) {
    __shared__ __align__(32) bf16 sWhh[Hh * Gg];   // 131072 B
    __shared__ __align__(32) bf16 sH[64 * Hh];     // 16384 B
    int tid = threadIdx.x, wid = tid >> 5, lane = tid & 31;
    int b0 = blockIdx.x * 64;
    int m0 = wid * 16;           // this wave's 16 rows within block tile
    int rtG = (b0 >> 4) + wid;   // this wave's global row tile
    int hi = lane >> 4, nn = lane & 15;

    {   // stage recurrent weights; zero h(0)
        const uint4* src = (const uint4*)whhT;
        uint4* dst = (uint4*)sWhh;
        for (int i = tid; i < (Hh * Gg) / 8; i += 128) dst[i] = src[i];
        uint4 z = make_uint4(0u, 0u, 0u, 0u);
        uint4* hz = (uint4*)sH;
        for (int i = tid; i < (64 * Hh) / 8; i += 128) hz[i] = z;
    }
    __syncthreads();

    float cReg[8][8];
#pragma unroll
    for (int g = 0; g < 8; ++g)
#pragma unroll
        for (int r = 0; r < 8; ++r) cReg[g][r] = 0.0f;

    for (int t = 0; t < Lseq; ++t) {
        int tt = reverse ? (Lseq - 1 - t) : t;
        bf16* yt = yout + ((size_t)tt * Bsz + b0) * D2 + dirOff;
        // base of this wave's frag-layout xp slab for step tt (32 tiles x 512B)
        const bf16* xpt = xp + (((size_t)tt * RT_N + rtG) * 32) * 256;

        // Prefetch next step's 16KB xp slab into cache (global_prefetch_b8).
        if (t + 1 < Lseq) {
            int tn = reverse ? (tt - 1) : (tt + 1);
            const char* nb = (const char*)(xp + (((size_t)tn * RT_N + rtG) * 32) * 256);
#pragma unroll
            for (int i = 0; i < 4; ++i)
                __builtin_prefetch((const void*)(nb + i * 4096 + lane * 128), 0, 1);
        }

        // Hoist all A fragments of h(t-1) before any h(t) is written.
        v16bf Af[4];
#pragma unroll
        for (int kt = 0; kt < 4; ++kt)
            Af[kt] = load_frag_a(sH, Hh, m0, kt * 32);

#pragma unroll
        for (int ct = 0; ct < 8; ++ct) {        // 16-wide column slice of H
            // Initialize accumulators with xp (input projection + biases).
            const bf16* xb = xpt + (size_t)lane * 8;
            v8f ai = cvt8(*(const v8bf*)(xb + (size_t)(ct +  0) * 256));
            v8f af = cvt8(*(const v8bf*)(xb + (size_t)(ct +  8) * 256));
            v8f ag = cvt8(*(const v8bf*)(xb + (size_t)(ct + 16) * 256));
            v8f ao = cvt8(*(const v8bf*)(xb + (size_t)(ct + 24) * 256));
#pragma unroll
            for (int kt = 0; kt < 4; ++kt) {    // K = 128 in chunks of 32
                v16bf Bi = load_frag_b(sWhh, Gg, kt * 32, ct * 16 + 0);
                v16bf Bf = load_frag_b(sWhh, Gg, kt * 32, ct * 16 + 128);
                v16bf Bg = load_frag_b(sWhh, Gg, kt * 32, ct * 16 + 256);
                v16bf Bo = load_frag_b(sWhh, Gg, kt * 32, ct * 16 + 384);
                ai = wmma_bf16(Af[kt], Bi, ai);
                af = wmma_bf16(Af[kt], Bf, af);
                ag = wmma_bf16(Af[kt], Bg, ag);
                ao = wmma_bf16(Af[kt], Bo, ao);
            }
            int j = ct * 16 + nn;               // hidden index 0..127
#pragma unroll
            for (int r = 0; r < 8; ++r) {
                int m = m0 + r + 8 * hi;        // local batch row
                float cn = sigf(af[r]) * cReg[ct][r] + sigf(ai[r]) * tanhfast(ag[r]);
                float hn = sigf(ao[r]) * tanhfast(cn);
                cReg[ct][r] = cn;
                bf16 hb = (bf16)hn;
                sH[m * Hh + j] = hb;                       // next-step A operand
                yt[(size_t)m * D2 + j] = hb;               // layer output
            }
        }
        // Same-wave DS store -> DS load ordering is guaranteed (LDS in-order);
        // compiler inserts s_wait_dscnt for the VGPR dependences.
    }
}

// ---------------------------------------------------------------------------
// Output head: out[b][o] = b_out[o] + sum_{t,n} relu(y[t][b][n]) * w_out[o][t*256+n]
// ---------------------------------------------------------------------------
__global__ __launch_bounds__(256)
void head_kernel(const bf16* __restrict__ y,      // [L][B][256]
                 const float* __restrict__ w_out, // [2][15360]
                 const float* __restrict__ b_out, // [2]
                 float* __restrict__ out) {       // [B][2]
    __shared__ float red0[256];
    __shared__ float red1[256];
    int b = blockIdx.x, n = threadIdx.x;
    float s0 = 0.0f, s1 = 0.0f;
    for (int t = 0; t < Lseq; ++t) {
        float v = (float)y[((size_t)t * Bsz + b) * D2 + n];
        v = fmaxf(v, 0.0f);
        int fi = t * D2 + n;
        s0 += v * w_out[fi];
        s1 += v * w_out[Lseq * D2 + fi];
    }
    red0[n] = s0; red1[n] = s1;
    __syncthreads();
    for (int s = 128; s > 0; s >>= 1) {
        if (n < s) { red0[n] += red0[n + s]; red1[n] += red1[n + s]; }
        __syncthreads();
    }
    if (n == 0) {
        out[(size_t)b * 2 + 0] = red0[0] + b_out[0];
        out[(size_t)b * 2 + 1] = red1[0] + b_out[1];
    }
}

// ---------------------------------------------------------------------------
// Host launcher. Workspace requirement: ~506 MB
//   (6x whhT + 4x wihT bf16, xp frag-layout bf16, two y ping-pong bf16 buffers)
// ---------------------------------------------------------------------------
extern "C" void kernel_launch(void* const* d_in, const int* in_sizes, int n_in,
                              void* d_out, int out_size, void* d_ws, size_t ws_size,
                              hipStream_t stream) {
    (void)in_sizes; (void)n_in; (void)out_size; (void)ws_size;

    const float* x = (const float*)d_in[0];
    auto W = [&](int layer, int dir, int which) -> const float* {
        return (const float*)d_in[1 + (layer * 2 + dir) * 4 + which];
    };
    const float* w_out = (const float*)d_in[25];
    const float* b_out = (const float*)d_in[26];

    char* p = (char*)d_ws;
    size_t off = 0;
    auto take = [&](size_t bytes) -> void* {
        void* r = p + off;
        off = (off + bytes + 255) & ~(size_t)255;
        return r;
    };
    bf16* whhT[6];
    for (int i = 0; i < 6; ++i) whhT[i] = (bf16*)take((size_t)Hh * Gg * 2);
    bf16* wihT[4];
    for (int i = 0; i < 4; ++i) wihT[i] = (bf16*)take((size_t)D2 * Gg * 2);
    bf16* xp = (bf16*)take((size_t)Lseq * Bsz * Gg * 2);
    bf16* yA = (bf16*)take((size_t)Lseq * Bsz * D2 * 2);
    bf16* yB = (bf16*)take((size_t)Lseq * Bsz * D2 * 2);

    // ---- weight prep (transpose + bf16 convert) ----
    for (int ld = 0; ld < 6; ++ld) {
        int layer = ld >> 1, dir = ld & 1;
        wtrans_kernel<<<(Hh * Gg + 255) / 256, 256, 0, stream>>>(
            W(layer, dir, 1), whhT[ld], Hh);
    }
    for (int i = 0; i < 4; ++i) {
        int layer = 1 + (i >> 1), dir = i & 1;
        wtrans_kernel<<<(D2 * Gg + 255) / 256, 256, 0, stream>>>(
            W(layer, dir, 0), wihT[i], D2);
    }

    const int xp0_blocks = (int)(((size_t)Lseq * Bsz * Gg) / 256);

    // ---- layer 0: x -> yA ----
    for (int dir = 0; dir < 2; ++dir) {
        xp0_kernel<<<xp0_blocks, 256, 0, stream>>>(
            x, W(0, dir, 0), W(0, dir, 2), W(0, dir, 3), xp);
        lstm_scan_kernel<<<Bsz / 64, 128, 0, stream>>>(
            xp, yA, whhT[dir], dir * Hh, dir);
    }
    // ---- layer 1: yA -> yB ----
    for (int dir = 0; dir < 2; ++dir) {
        xp_gemm_kernel<<<dim3(Bsz / 64, Lseq), 256, 0, stream>>>(
            yA, wihT[dir], W(1, dir, 2), W(1, dir, 3), xp);
        lstm_scan_kernel<<<Bsz / 64, 128, 0, stream>>>(
            xp, yB, whhT[2 + dir], dir * Hh, dir);
    }
    // ---- layer 2: yB -> yA ----
    for (int dir = 0; dir < 2; ++dir) {
        xp_gemm_kernel<<<dim3(Bsz / 64, Lseq), 256, 0, stream>>>(
            yB, wihT[2 + dir], W(2, dir, 2), W(2, dir, 3), xp);
        lstm_scan_kernel<<<Bsz / 64, 128, 0, stream>>>(
            xp, yA, whhT[4 + dir], dir * Hh, dir);
    }
    // ---- head ----
    head_kernel<<<Bsz, 256, 0, stream>>>(yA, w_out, b_out, (float*)d_out);
}
